// PyGEGNNDecoder_81535659147810
// MI455X (gfx1250) — compile-verified, matrix operand order checked by reading, not run on hardware
//
#include <hip/hip_runtime.h>
#include <hip/hip_bf16.h>
#include <stddef.h>

// ---------------------------------------------------------------------------
// EGNN decoder forward for MI455X (gfx1250, wave32, WMMA f32_16x16x32_f16)
// ---------------------------------------------------------------------------

typedef _Float16 half_t;
typedef __attribute__((ext_vector_type(2)))  _Float16 h2v;
typedef __attribute__((ext_vector_type(16))) _Float16 v16h;
typedef __attribute__((ext_vector_type(8)))  float    v8f;

#define EGNN_N 8192
#define EGNN_E 262144
#define EGNN_H 128
#define EGNN_L 6
#define S_STRIDE 264   // 256 data halfs + 8 pad (bank-conflict relief), 16B-aligned rows
#define WPB 4          // waves per block in tile kernels (128 threads)

// Fast SiLU: x * v_rcp_f32(1 + exp(-x)).  Avoids the IEEE-correct division
// expansion (v_div_scale/v_div_fmas chain); v_exp/v_rcp are TRANS ops that
// co-execute with WMMA on CDNA5.
__device__ __forceinline__ float silu_f(float x) {
  return x * __builtin_amdgcn_rcpf(1.0f + __expf(-x));
}

// Build a 16x32 f16 A-fragment (wave32 layout, CDNA5 ISA §7.12.2)
// from an LDS tile: row M = lane&15, per-lane K pattern depends on lane>>4.
__device__ __forceinline__ v16h frag_from_lds(const half_t* base, int stride) {
  const int lane = threadIdx.x & 31;
  const int M = lane & 15;
  const int hf = lane >> 4;
  const half_t* p = base + M * stride + hf * 8;
  v16h a;
#pragma unroll
  for (int v = 0; v < 8; ++v) {
    const int k = 2 * v + ((v >= 4) ? 8 : 0);
    h2v t = *(const h2v*)(p + k);
    a[2 * v]     = t[0];
    a[2 * v + 1] = t[1];
  }
  return a;
}

// Build a 32x16 f16 B-fragment from a pre-transposed weight Wt[N][Kdim]
// (column N per lane, same per-lane K pattern as A).
__device__ __forceinline__ v16h frag_B_from_global(const half_t* __restrict__ Wt,
                                                   int Kdim, int n0, int k0) {
  const int lane = threadIdx.x & 31;
  const int N = lane & 15;
  const int hf = lane >> 4;
  const half_t* p = Wt + (size_t)(n0 + N) * Kdim + k0 + hf * 8;
  v16h b;
#pragma unroll
  for (int v = 0; v < 8; ++v) {
    const int k = 2 * v + ((v >= 4) ? 8 : 0);
    h2v t = *(const h2v*)(p + k);
    b[2 * v]     = t[0];
    b[2 * v + 1] = t[1];
  }
  return b;
}

__device__ __forceinline__ v8f wmma16(v16h a, v16h b, v8f c) {
  return __builtin_amdgcn_wmma_f32_16x16x32_f16(
      /*neg_a=*/false, a, /*neg_b=*/false, b,
      /*c_mod=*/(short)0, c, /*reuse_a=*/false, /*reuse_b=*/false);
}

// ---------------------------------------------------------------------------
// Utility kernels
// ---------------------------------------------------------------------------

__global__ void egnn_zero_kernel(float* __restrict__ p, int n) {
  const int i = blockIdx.x * blockDim.x + threadIdx.x;
  if (i < n) p[i] = 0.0f;
}

__global__ void egnn_deg_kernel(const int* __restrict__ erow, float* __restrict__ deg) {
  const int e = blockIdx.x * blockDim.x + threadIdx.x;
  if (e < EGNN_E) atomicAdd(&deg[erow[e]], 1.0f);
}

// W[Kdim][Ndim] (f32, row-major) -> Wt[Ndim][KPad] (f16), zero-padded in K.
__global__ void egnn_tr_kernel(const float* __restrict__ W, half_t* __restrict__ Wt,
                               int Kdim, int Ndim, int KPad) {
  const int i = blockIdx.x * blockDim.x + threadIdx.x;
  if (i >= Ndim * KPad) return;
  const int n = i / KPad;
  const int k = i - n * KPad;
  Wt[i] = (k < Kdim) ? (half_t)W[(size_t)k * Ndim + n] : (half_t)0.0f;
}

// ---------------------------------------------------------------------------
// Latent injection: h = (concat(atom,z) @ W1).silu @ W2).silu @ W3   [N,128]
// ---------------------------------------------------------------------------

__global__ __launch_bounds__(128) void egnn_inject_kernel(
    const float* __restrict__ atom_types, const float* __restrict__ z,
    const half_t* __restrict__ W1t, const float* __restrict__ b1,   // [256][96]
    const half_t* __restrict__ W2t, const float* __restrict__ b2,   // [128][256]
    const half_t* __restrict__ W3t, const float* __restrict__ b3,   // [128][128]
    float* __restrict__ h, half_t* __restrict__ h16)
{
  __shared__ __align__(16) half_t smem[WPB][16][S_STRIDE];
  const int wave = threadIdx.x >> 5;
  const int lane = threadIdx.x & 31;
  const int ln = lane & 15;
  const int Mbase = (lane >> 4) * 8;
  half_t (*s)[S_STRIDE] = smem[wave];
  const int n0node = (blockIdx.x * WPB + wave) * 16;

  // Stage input rows [atom(16) | z(64) | pad(16)] -> 16 x 96 f16 in LDS.
#pragma unroll
  for (int it = 0; it < 12; ++it) {
    const int f = it * 32 + lane;      // 384 chunks of 4 halfs
    const int ni = f / 24;
    const int w = f - ni * 24;
    const int node = n0node + ni;
    float v[4];
#pragma unroll
    for (int j = 0; j < 4; ++j) {
      const int k = w * 4 + j;
      float val = 0.0f;
      if (k < 16)      val = atom_types[(size_t)node * 16 + k];
      else if (k < 80) val = z[(size_t)(node >> 5) * 64 + (k - 16)];
      v[j] = val;
    }
    h2v* d = (h2v*)&s[ni][w * 4];
    h2v p0; p0[0] = (half_t)v[0]; p0[1] = (half_t)v[1]; d[0] = p0;
    h2v p1; p1[0] = (half_t)v[2]; p1[1] = (half_t)v[3]; d[1] = p1;
  }
  __syncthreads();

  v16h Az[3];
#pragma unroll
  for (int c = 0; c < 3; ++c) Az[c] = frag_from_lds(&s[0][c * 32], S_STRIDE);
  __syncthreads();

  // Layer 1: K=96 -> 256 out, SiLU, store to LDS cols 0..255
#pragma unroll
  for (int t = 0; t < 16; ++t) {
    const int n = t * 16 + ln;
    const float bb = b1[n];
    v16h Bf[3];
#pragma unroll
    for (int c = 0; c < 3; ++c) Bf[c] = frag_B_from_global(W1t, 96, t * 16, c * 32);
    v8f acc;
#pragma unroll
    for (int r = 0; r < 8; ++r) acc[r] = bb;
#pragma unroll
    for (int c = 0; c < 3; ++c) acc = wmma16(Az[c], Bf[c], acc);
#pragma unroll
    for (int r = 0; r < 8; ++r)
      s[Mbase + r][n] = (half_t)silu_f(acc[r]);
  }
  __syncthreads();

  v16h A1[8];
#pragma unroll
  for (int c = 0; c < 8; ++c) A1[c] = frag_from_lds(&s[0][c * 32], S_STRIDE);
  __syncthreads();

  // Layer 2: K=256 -> 128 out, SiLU, store to LDS cols 0..127
#pragma unroll
  for (int t = 0; t < 8; ++t) {
    const int n = t * 16 + ln;
    const float bb = b2[n];
    v16h Bf[8];
#pragma unroll
    for (int c = 0; c < 8; ++c) Bf[c] = frag_B_from_global(W2t, 256, t * 16, c * 32);
    v8f acc;
#pragma unroll
    for (int r = 0; r < 8; ++r) acc[r] = bb;
#pragma unroll
    for (int c = 0; c < 8; ++c) acc = wmma16(A1[c], Bf[c], acc);
#pragma unroll
    for (int r = 0; r < 8; ++r)
      s[Mbase + r][n] = (half_t)silu_f(acc[r]);
  }
  __syncthreads();

  v16h A2[4];
#pragma unroll
  for (int c = 0; c < 4; ++c) A2[c] = frag_from_lds(&s[0][c * 32], S_STRIDE);

  // Layer 3: K=128 -> 128 out, linear -> global h (f32) + h16 (f16)
#pragma unroll
  for (int t = 0; t < 8; ++t) {
    const int n = t * 16 + ln;
    const float bb = b3[n];
    v16h Bf[4];
#pragma unroll
    for (int c = 0; c < 4; ++c) Bf[c] = frag_B_from_global(W3t, 128, t * 16, c * 32);
    v8f acc;
#pragma unroll
    for (int r = 0; r < 8; ++r) acc[r] = bb;
#pragma unroll
    for (int c = 0; c < 4; ++c) acc = wmma16(A2[c], Bf[c], acc);
#pragma unroll
    for (int r = 0; r < 8; ++r) {
      const size_t idx = (size_t)(n0node + Mbase + r) * EGNN_H + n;
      h[idx] = acc[r];
      h16[idx] = (half_t)acc[r];
    }
  }
}

// ---------------------------------------------------------------------------
// Edge kernel: one wave handles 16 edges.
//   m  = silu(silu([h_r|h_c|d2] @ eW1 + eb1) @ eW2 + eb2)
//   cw = silu(m @ cW1 + cb1) @ cW2
//   scatter-add:  agg[row] += m;  coordu[row] += cw * dpos
// ---------------------------------------------------------------------------

__global__ __launch_bounds__(128) void egnn_edge_kernel(
    const float* __restrict__ pos, const half_t* __restrict__ h16,
    const int* __restrict__ erow, const int* __restrict__ ecol,
    const half_t* __restrict__ eW1t, const float* __restrict__ eW1last,
    const float* __restrict__ eb1,
    const half_t* __restrict__ eW2t, const float* __restrict__ eb2,
    const half_t* __restrict__ cW1t, const float* __restrict__ cb1,
    const float* __restrict__ cW2,
    float* __restrict__ agg, float* __restrict__ coordu)
{
  __shared__ __align__(16) half_t smem[WPB][16][S_STRIDE];
  const int wave = threadIdx.x >> 5;
  const int lane = threadIdx.x & 31;
  const int ln = lane & 15;
  const int Mbase = (lane >> 4) * 8;
  half_t (*s)[S_STRIDE] = smem[wave];
  const int e0 = (blockIdx.x * WPB + wave) * 16;

  const int eL = e0 + ln;
  const int rL = erow[eL];
  const int cL = ecol[eL];
  const float dx = pos[rL * 3 + 0] - pos[cL * 3 + 0];
  const float dy = pos[rL * 3 + 1] - pos[cL * 3 + 1];
  const float dz = pos[rL * 3 + 2] - pos[cL * 3 + 2];
  const float dsq = dx * dx + dy * dy + dz * dz;

  // Per-row broadcasts hoisted out of the N-tile loops (8 permutes each).
  float drow[8];
  int   rowM[8];
#pragma unroll
  for (int r = 0; r < 8; ++r) {
    drow[r] = __shfl(dsq, Mbase + r, 32);
    rowM[r] = __shfl(rL,  Mbase + r, 32);
  }

  // Stage ef = [h16[row] | h16[col]] : 16 edges x 256 f16, 16B per lane-iter.
#pragma unroll
  for (int it = 0; it < 16; ++it) {
    const int f = it * 32 + lane;       // 512 chunks of 8 halfs
    const int ei = f >> 5;              // edge within tile
    const int w = f & 31;               // 16B chunk within the 512B row
    const int node = (w < 16) ? __shfl(rL, ei, 32) : __shfl(cL, ei, 32);
    const uint4 v = *(const uint4*)(h16 + (size_t)node * EGNN_H + (size_t)(w & 15) * 8);
    *(uint4*)(&s[ei][w * 8]) = v;
  }
  __syncthreads();

  v16h Aef[8];
#pragma unroll
  for (int c = 0; c < 8; ++c) Aef[c] = frag_from_lds(&s[0][c * 32], S_STRIDE);
  __syncthreads();

  // Edge MLP layer 1 (K=256 GEMM + rank-1 dist term) + SiLU -> m1 (cols 0..127)
#pragma unroll
  for (int t = 0; t < 8; ++t) {
    const int n = t * 16 + ln;
    const float bb = eb1[n];
    const float wl = eW1last[n];
    v16h Bf[8];
#pragma unroll
    for (int c = 0; c < 8; ++c) Bf[c] = frag_B_from_global(eW1t, 256, t * 16, c * 32);
    v8f acc;
#pragma unroll
    for (int r = 0; r < 8; ++r) acc[r] = bb;
#pragma unroll
    for (int c = 0; c < 8; ++c) acc = wmma16(Aef[c], Bf[c], acc);
#pragma unroll
    for (int r = 0; r < 8; ++r)
      s[Mbase + r][n] = (half_t)silu_f(acc[r] + drow[r] * wl);
  }
  __syncthreads();

  v16h Am[4];
#pragma unroll
  for (int c = 0; c < 4; ++c) Am[c] = frag_from_lds(&s[0][c * 32], S_STRIDE);
  __syncthreads();

  // Edge MLP layer 2 + SiLU -> m (cols 128..255) + scatter-add into agg
#pragma unroll
  for (int t = 0; t < 8; ++t) {
    const int n = t * 16 + ln;
    const float bb = eb2[n];
    v16h Bf[4];
#pragma unroll
    for (int c = 0; c < 4; ++c) Bf[c] = frag_B_from_global(eW2t, 128, t * 16, c * 32);
    v8f acc;
#pragma unroll
    for (int r = 0; r < 8; ++r) acc[r] = bb;
#pragma unroll
    for (int c = 0; c < 4; ++c) acc = wmma16(Am[c], Bf[c], acc);
#pragma unroll
    for (int r = 0; r < 8; ++r) {
      const float x = silu_f(acc[r]);
      s[Mbase + r][128 + n] = (half_t)x;
      atomicAdd(&agg[(size_t)rowM[r] * EGNN_H + n], x);
    }
  }
  __syncthreads();

  v16h Am2[4];
#pragma unroll
  for (int c = 0; c < 4; ++c) Am2[c] = frag_from_lds(&s[0][128 + c * 32], S_STRIDE);
  __syncthreads();

  // Coord MLP layer 1 + SiLU -> t3 (cols 0..127)
#pragma unroll
  for (int t = 0; t < 8; ++t) {
    const int n = t * 16 + ln;
    const float bb = cb1[n];
    v16h Bf[4];
#pragma unroll
    for (int c = 0; c < 4; ++c) Bf[c] = frag_B_from_global(cW1t, 128, t * 16, c * 32);
    v8f acc;
#pragma unroll
    for (int r = 0; r < 8; ++r) acc[r] = bb;
#pragma unroll
    for (int c = 0; c < 4; ++c) acc = wmma16(Am2[c], Bf[c], acc);
#pragma unroll
    for (int r = 0; r < 8; ++r)
      s[Mbase + r][n] = (half_t)silu_f(acc[r]);
  }
  __syncthreads();

  // cw = t3 . cW2 (per edge; half the K per lane pair, combine with shfl_xor)
  float cw = 0.0f;
  const int kbase = (lane >> 4) * 64;
#pragma unroll
  for (int k = 0; k < 64; k += 2) {
    const h2v t = *(const h2v*)&s[ln][kbase + k];
    cw += (float)t[0] * cW2[kbase + k] + (float)t[1] * cW2[kbase + k + 1];
  }
  cw += __shfl_xor(cw, 16, 32);

  if (lane < 16) {
    atomicAdd(&coordu[(size_t)rL * 3 + 0], cw * dx);
    atomicAdd(&coordu[(size_t)rL * 3 + 1], cw * dy);
    atomicAdd(&coordu[(size_t)rL * 3 + 2], cw * dz);
  }
}

// ---------------------------------------------------------------------------
// Node kernel: nu = silu([h|agg] @ nW1 + nb1) @ nW2 + nb2 ; htmp = h + nu
// ---------------------------------------------------------------------------

__global__ __launch_bounds__(128) void egnn_node_kernel(
    const float* __restrict__ h, const float* __restrict__ agg,
    const half_t* __restrict__ nW1t, const float* __restrict__ nb1,
    const half_t* __restrict__ nW2t, const float* __restrict__ nb2,
    float* __restrict__ htmp)
{
  __shared__ __align__(16) half_t smem[WPB][16][S_STRIDE];
  const int wave = threadIdx.x >> 5;
  const int lane = threadIdx.x & 31;
  const int ln = lane & 15;
  const int Mbase = (lane >> 4) * 8;
  half_t (*s)[S_STRIDE] = smem[wave];
  const int n0node = (blockIdx.x * WPB + wave) * 16;

  // Stage x = [h | agg] : 16 nodes x 256 halfs (f32 -> f16 on the fly).
#pragma unroll
  for (int it = 0; it < 32; ++it) {
    const int f = it * 32 + lane;       // 1024 chunks of 4 halfs
    const int ni = f >> 6;
    const int w = f & 63;
    const float4 v = (w < 32)
        ? *(const float4*)(h   + (size_t)(n0node + ni) * EGNN_H + (size_t)w * 4)
        : *(const float4*)(agg + (size_t)(n0node + ni) * EGNN_H + (size_t)(w - 32) * 4);
    h2v* d = (h2v*)&s[ni][w * 4];
    h2v p0; p0[0] = (half_t)v.x; p0[1] = (half_t)v.y; d[0] = p0;
    h2v p1; p1[0] = (half_t)v.z; p1[1] = (half_t)v.w; d[1] = p1;
  }
  __syncthreads();

  v16h Ax[8];
#pragma unroll
  for (int c = 0; c < 8; ++c) Ax[c] = frag_from_lds(&s[0][c * 32], S_STRIDE);
  __syncthreads();

  // Node MLP layer 1 (K=256) + SiLU -> LDS cols 0..127
#pragma unroll
  for (int t = 0; t < 8; ++t) {
    const int n = t * 16 + ln;
    const float bb = nb1[n];
    v16h Bf[8];
#pragma unroll
    for (int c = 0; c < 8; ++c) Bf[c] = frag_B_from_global(nW1t, 256, t * 16, c * 32);
    v8f acc;
#pragma unroll
    for (int r = 0; r < 8; ++r) acc[r] = bb;
#pragma unroll
    for (int c = 0; c < 8; ++c) acc = wmma16(Ax[c], Bf[c], acc);
#pragma unroll
    for (int r = 0; r < 8; ++r)
      s[Mbase + r][n] = (half_t)silu_f(acc[r]);
  }
  __syncthreads();

  v16h At[4];
#pragma unroll
  for (int c = 0; c < 4; ++c) At[c] = frag_from_lds(&s[0][c * 32], S_STRIDE);

  // Node MLP layer 2 (K=128), residual add -> htmp (f32)
#pragma unroll
  for (int t = 0; t < 8; ++t) {
    const int n = t * 16 + ln;
    const float bb = nb2[n];
    v16h Bf[4];
#pragma unroll
    for (int c = 0; c < 4; ++c) Bf[c] = frag_B_from_global(nW2t, 128, t * 16, c * 32);
    v8f acc;
#pragma unroll
    for (int r = 0; r < 8; ++r) acc[r] = bb;
#pragma unroll
    for (int c = 0; c < 4; ++c) acc = wmma16(At[c], Bf[c], acc);
#pragma unroll
    for (int r = 0; r < 8; ++r) {
      const size_t idx = (size_t)(n0node + Mbase + r) * EGNN_H + n;
      htmp[idx] = h[idx] + acc[r];
    }
  }
}

// ---------------------------------------------------------------------------
// LayerNorm + position update: one wave per node (wave32 shfl reductions).
// ---------------------------------------------------------------------------

__global__ __launch_bounds__(256) void egnn_ln_pos_kernel(
    const float* __restrict__ htmp, const float* __restrict__ g,
    const float* __restrict__ bb, const float* __restrict__ coordu,
    const float* __restrict__ deg, float* __restrict__ h,
    half_t* __restrict__ h16, float* __restrict__ pos)
{
  const int wave = threadIdx.x >> 5;
  const int lane = threadIdx.x & 31;
  const int node = blockIdx.x * 8 + wave;
  const float* x = htmp + (size_t)node * EGNN_H;

  float v[4];
  float sum = 0.0f;
#pragma unroll
  for (int j = 0; j < 4; ++j) { v[j] = x[lane + 32 * j]; sum += v[j]; }
#pragma unroll
  for (int o = 16; o >= 1; o >>= 1) sum += __shfl_xor(sum, o, 32);
  const float mean = sum * (1.0f / 128.0f);

  float vs = 0.0f;
#pragma unroll
  for (int j = 0; j < 4; ++j) { const float d = v[j] - mean; vs += d * d; }
#pragma unroll
  for (int o = 16; o >= 1; o >>= 1) vs += __shfl_xor(vs, o, 32);
  const float rstd = rsqrtf(vs * (1.0f / 128.0f) + 1e-5f);

#pragma unroll
  for (int j = 0; j < 4; ++j) {
    const int k = lane + 32 * j;
    const float y = (v[j] - mean) * rstd * g[k] + bb[k];
    h[(size_t)node * EGNN_H + k] = y;
    h16[(size_t)node * EGNN_H + k] = (half_t)y;
  }
  if (lane < 3) {
    pos[node * 3 + lane] += coordu[node * 3 + lane] *
                            __builtin_amdgcn_rcpf(deg[node] + 1e-6f);
  }
}

// ---------------------------------------------------------------------------
// Host-side orchestration
// ---------------------------------------------------------------------------

extern "C" void kernel_launch(void* const* d_in, const int* in_sizes, int n_in,
                              void* d_out, int out_size, void* d_ws, size_t ws_size,
                              hipStream_t stream) {
  (void)in_sizes; (void)n_in; (void)out_size; (void)ws_size;

  const float* z        = (const float*)d_in[0];
  const float* atom     = (const float*)d_in[1];
  const float* pos_init = (const float*)d_in[2];
  const int*   eidx     = (const int*)  d_in[3];
  /* d_in[4] = batch (unused: z repeat uses node>>5 since A=32) */
  const float* inj_W1 = (const float*)d_in[5];
  const float* inj_b1 = (const float*)d_in[6];
  const float* inj_W2 = (const float*)d_in[7];
  const float* inj_b2 = (const float*)d_in[8];
  const float* inj_W3 = (const float*)d_in[9];
  const float* inj_b3 = (const float*)d_in[10];
  const float* edge_W1 = (const float*)d_in[11];
  const float* edge_b1 = (const float*)d_in[12];
  const float* edge_W2 = (const float*)d_in[13];
  const float* edge_b2 = (const float*)d_in[14];
  const float* node_W1 = (const float*)d_in[15];
  const float* node_b1 = (const float*)d_in[16];
  const float* node_W2 = (const float*)d_in[17];
  const float* node_b2 = (const float*)d_in[18];
  const float* coord_W1 = (const float*)d_in[19];
  const float* coord_b1 = (const float*)d_in[20];
  const float* coord_W2 = (const float*)d_in[21];
  const float* ln_g = (const float*)d_in[22];
  const float* ln_b = (const float*)d_in[23];

  const int* erow = eidx;
  const int* ecol = eidx + EGNN_E;

  // Workspace carve-up (256B aligned)
  char* wsp = (char*)d_ws;
  size_t off = 0;
  auto alloc = [&](size_t bytes) -> void* {
    void* p = (void*)(wsp + off);
    off += (bytes + 255) & ~(size_t)255;
    return p;
  };
  float*  h      = (float*)alloc((size_t)EGNN_N * EGNN_H * 4);
  float*  htmp   = (float*)alloc((size_t)EGNN_N * EGNN_H * 4);
  float*  agg    = (float*)alloc((size_t)EGNN_N * EGNN_H * 4);
  float*  coordu = (float*)alloc((size_t)EGNN_N * 3 * 4);
  float*  deg    = (float*)alloc((size_t)EGNN_N * 4);
  half_t* h16    = (half_t*)alloc((size_t)EGNN_N * EGNN_H * 2);
  half_t* eW1t   = (half_t*)alloc((size_t)EGNN_L * 128 * 256 * 2);
  half_t* eW2t   = (half_t*)alloc((size_t)EGNN_L * 128 * 128 * 2);
  half_t* nW1t   = (half_t*)alloc((size_t)EGNN_L * 128 * 256 * 2);
  half_t* nW2t   = (half_t*)alloc((size_t)EGNN_L * 128 * 128 * 2);
  half_t* cW1t   = (half_t*)alloc((size_t)EGNN_L * 128 * 128 * 2);
  half_t* iW1t   = (half_t*)alloc((size_t)256 * 96 * 2);
  half_t* iW2t   = (half_t*)alloc((size_t)128 * 256 * 2);
  half_t* iW3t   = (half_t*)alloc((size_t)128 * 128 * 2);

  // Weight transpose/convert to f16 [N][K] layout.
  for (int l = 0; l < EGNN_L; ++l) {
    egnn_tr_kernel<<<(128 * 256 + 255) / 256, 256, 0, stream>>>(
        edge_W1 + (size_t)l * 257 * 128, eW1t + (size_t)l * 128 * 256, 256, 128, 256);
    egnn_tr_kernel<<<(128 * 128 + 255) / 256, 256, 0, stream>>>(
        edge_W2 + (size_t)l * 128 * 128, eW2t + (size_t)l * 128 * 128, 128, 128, 128);
    egnn_tr_kernel<<<(128 * 256 + 255) / 256, 256, 0, stream>>>(
        node_W1 + (size_t)l * 256 * 128, nW1t + (size_t)l * 128 * 256, 256, 128, 256);
    egnn_tr_kernel<<<(128 * 128 + 255) / 256, 256, 0, stream>>>(
        node_W2 + (size_t)l * 128 * 128, nW2t + (size_t)l * 128 * 128, 128, 128, 128);
    egnn_tr_kernel<<<(128 * 128 + 255) / 256, 256, 0, stream>>>(
        coord_W1 + (size_t)l * 128 * 128, cW1t + (size_t)l * 128 * 128, 128, 128, 128);
  }
  egnn_tr_kernel<<<(256 * 96 + 255) / 256, 256, 0, stream>>>(inj_W1, iW1t, 80, 256, 96);
  egnn_tr_kernel<<<(128 * 256 + 255) / 256, 256, 0, stream>>>(inj_W2, iW2t, 256, 128, 256);
  egnn_tr_kernel<<<(128 * 128 + 255) / 256, 256, 0, stream>>>(inj_W3, iW3t, 128, 128, 128);

  // Degree
  egnn_zero_kernel<<<(EGNN_N + 255) / 256, 256, 0, stream>>>(deg, EGNN_N);
  egnn_deg_kernel<<<(EGNN_E + 255) / 256, 256, 0, stream>>>(erow, deg);

  // pos lives in d_out, updated in place across layers.
  hipMemcpyAsync(d_out, pos_init, (size_t)EGNN_N * 3 * 4, hipMemcpyDeviceToDevice, stream);
  float* pos = (float*)d_out;

  // Latent injection -> h, h16
  egnn_inject_kernel<<<EGNN_N / (16 * WPB), 128, 0, stream>>>(
      atom, z, iW1t, inj_b1, iW2t, inj_b2, iW3t, inj_b3, h, h16);

  for (int l = 0; l < EGNN_L; ++l) {
    egnn_zero_kernel<<<(EGNN_N * EGNN_H + 255) / 256, 256, 0, stream>>>(agg, EGNN_N * EGNN_H);
    egnn_zero_kernel<<<(EGNN_N * 3 + 255) / 256, 256, 0, stream>>>(coordu, EGNN_N * 3);

    egnn_edge_kernel<<<EGNN_E / (16 * WPB), 128, 0, stream>>>(
        pos, h16, erow, ecol,
        eW1t + (size_t)l * 128 * 256,
        edge_W1 + (size_t)l * 257 * 128 + 256 * 128,   // dist_sq row of eW1 (f32)
        edge_b1 + l * 128,
        eW2t + (size_t)l * 128 * 128, edge_b2 + l * 128,
        cW1t + (size_t)l * 128 * 128, coord_b1 + l * 128,
        coord_W2 + l * 128,
        agg, coordu);

    egnn_node_kernel<<<EGNN_N / (16 * WPB), 128, 0, stream>>>(
        h, agg,
        nW1t + (size_t)l * 128 * 256, node_b1 + l * 128,
        nW2t + (size_t)l * 128 * 128, node_b2 + l * 128,
        htmp);

    egnn_ln_pos_kernel<<<EGNN_N / 8, 256, 0, stream>>>(
        htmp, ln_g + l * 128, ln_b + l * 128, coordu, deg, h, h16, pos);
  }
}
